// ViT_42588895708020
// MI455X (gfx1250) — matrix operand order repaired and checked by reading, not compiled, not run on hardware
//
#include <hip/hip_runtime.h>
#include <hip/hip_bf16.h>
#include <math.h>

// ---------------- model dims ----------------
#define BATCH 512
#define NTOK  65          // cls + 8x8 patches
#define DIMD  384
#define HEADS 6
#define DHEAD 64
#define DEPTH 12
#define MLPD  1536
#define NCLS  1000
#define PDIM  768         // 16*16*3
#define INNER (HEADS*DHEAD)   // 384
#define QKVW  (3*INNER)       // 1152
#define TOPK  16

// ---------------- vector types ----------------
typedef __attribute__((ext_vector_type(16))) _Float16 v16h;
typedef __attribute__((ext_vector_type(8)))  _Float16 v8h;
typedef __attribute__((ext_vector_type(4)))  _Float16 v4h;
typedef __attribute__((ext_vector_type(8)))  float    v8f;
typedef __attribute__((ext_vector_type(4)))  int      i32x4;
// pointer-to-AS-qualified-int4 types for the async-LDS builtin
typedef __attribute__((address_space(1))) i32x4* as1_i32x4_p;
typedef __attribute__((address_space(3))) i32x4* as3_i32x4_p;

union Frag16 { v16h v; _Float16 e[16]; };

__device__ inline v8f wmma32(const v16h& a, const v16h& b, const v8f& c) {
  return __builtin_amdgcn_wmma_f32_16x16x32_f16(false, a, false, b, (short)0, c, false, false);
}

// async global->LDS (CDNA5) if the toolchain exposes it
#if defined(__has_builtin)
#  if __has_builtin(__builtin_amdgcn_global_load_async_to_lds_b128)
#    define HAVE_ASYNC_LDS 1
#  endif
#endif
#ifndef HAVE_ASYNC_LDS
#  define HAVE_ASYNC_LDS 0
#endif

// ---------------- generic WMMA GEMM ----------------
// C[M,Nc] = A[M,K](f32) * Wt[Nc,K](f16, pre-transposed)  (+bias)(gelu)(+res)
// Block tile 128x128, BK=32, 256 threads = 8 waves (4 along M x 2 along N),
// wave tile 32x64 -> 2x4 fragments -> 8 v_wmma per K-step.
// Requires: M % 128 == 0, K % 32 == 0. NGUARD=true adds column bounds (head).
#define BM 128
#define BN 128
#define BKK 32
#define FLAG_BIAS 1
#define FLAG_GELU 2
#define FLAG_RES  4

template <bool NGUARD>
__global__ __launch_bounds__(256)
void gemm_wmma_kernel(const float* __restrict__ A, const _Float16* __restrict__ Wt,
                      const float* __restrict__ bias, const float* __restrict__ resid,
                      float* __restrict__ Cm, int M, int K, int Nc, int flags) {
  __shared__ __align__(16) _Float16 As[BM][BKK + 8];   // [m][k]
  __shared__ __align__(16) _Float16 Bs[BN][BKK + 8];   // [n][k]
  const int tid  = threadIdx.x;
  const int lane = tid & 31;
  const int wave = tid >> 5;
  const int wm   = wave & 3;        // 4 waves along M (32 rows each)
  const int wn   = wave >> 2;       // 2 waves along N (64 cols each)
  const int row0 = blockIdx.x * BM;
  const int col0 = blockIdx.y * BN;
  const int llo  = lane & 15;
  const int lhi  = lane >> 4;

  v8f zero = {};
  v8f acc[2][4];
#pragma unroll
  for (int r = 0; r < 2; ++r)
#pragma unroll
    for (int c = 0; c < 4; ++c) acc[r][c] = zero;

  for (int k0 = 0; k0 < K; k0 += BKK) {
    // ---- stage A tile 128x32: f32 -> f16 ----
    // Phase 1: issue all 4 b128 loads back-to-back (distinct registers),
    // Phase 2: convert + ds_store, so waits are staggered not serialized.
    float4 fq[4];
#pragma unroll
    for (int t = 0; t < 4; ++t) {
      int e  = tid + t * 256;          // quad index over 1024
      int m  = e >> 3;                 // 8 quads per 32-wide row
      int k4 = (e & 7) << 2;
      fq[t] = *reinterpret_cast<const float4*>(&A[(size_t)(row0 + m) * K + k0 + k4]);
    }
    // ---- stage B tile 128x32 from pre-packed f16 Wt[Nc][K]: async copy ----
#pragma unroll
    for (int t = 0; t < 2; ++t) {
      int e  = tid + t * 256;          // 16-byte chunk index over 512
      int n  = e >> 2;                 // 4 chunks per 32-half row
      int kc = (e & 3) << 3;
      const _Float16* gp = &Wt[(size_t)(col0 + n) * K + k0 + kc];
      _Float16* lp = &Bs[n][kc];
      if (!NGUARD || (col0 + n) < Nc) {
#if HAVE_ASYNC_LDS
        __builtin_amdgcn_global_load_async_to_lds_b128(
            (as1_i32x4_p)(uintptr_t)gp, (as3_i32x4_p)(uintptr_t)lp, 0, 0);
#else
        *reinterpret_cast<v8h*>(lp) = *reinterpret_cast<const v8h*>(gp);
#endif
      }
    }
    // prefetch next A K-tile (global_prefetch_b8)
    if (k0 + BKK < K) {
      __builtin_prefetch(&A[(size_t)(row0 + (tid >> 1)) * K + k0 + BKK + (tid & 1) * 16], 0, 1);
    }
    // Phase 2: convert the A quads and store to LDS
#pragma unroll
    for (int t = 0; t < 4; ++t) {
      int e  = tid + t * 256;
      int m  = e >> 3;
      int k4 = (e & 7) << 2;
      v4h hv = {(_Float16)fq[t].x, (_Float16)fq[t].y, (_Float16)fq[t].z, (_Float16)fq[t].w};
      *reinterpret_cast<v4h*>(&As[m][k4]) = hv;
    }
#if HAVE_ASYNC_LDS
#  if __has_builtin(__builtin_amdgcn_s_wait_asynccnt)
    __builtin_amdgcn_s_wait_asynccnt(0);
#  else
    asm volatile("s_wait_asynccnt 0x0" ::: "memory");
#  endif
#endif
    __syncthreads();

    // ---- fragments: 2 A-frags, 4 B-frags, 8 WMMAs ----
    Frag16 a[2], bf[4];
#pragma unroll
    for (int r = 0; r < 2; ++r) {
      const _Float16* ap = &As[wm * 32 + r * 16 + llo][lhi * 16];
#pragma unroll
      for (int hh = 0; hh < 8; ++hh) {
        a[r].e[hh]     = ap[hh];        // K = lhi*16 + hh
        a[r].e[8 + hh] = ap[8 + hh];    // K = lhi*16 + 8 + hh
      }
    }
#pragma unroll
    for (int c = 0; c < 4; ++c) {
      const _Float16* bp = &Bs[wn * 64 + c * 16 + llo][lhi * 16];
#pragma unroll
      for (int hh = 0; hh < 16; ++hh) bf[c].e[hh] = bp[hh];  // K = lhi*16 + hh
    }
#pragma unroll
    for (int r = 0; r < 2; ++r)
#pragma unroll
      for (int c = 0; c < 4; ++c)
        acc[r][c] = wmma32(a[r].v, bf[c].v, acc[r][c]);
    __syncthreads();
  }

  // ---- epilogue: row = 16r + 8*(lane>>4) + i, col = 16c + (lane&15) ----
#pragma unroll
  for (int r = 0; r < 2; ++r) {
#pragma unroll
    for (int c = 0; c < 4; ++c) {
      int gcol = col0 + wn * 64 + c * 16 + llo;
      if (NGUARD && gcol >= Nc) continue;
#pragma unroll
      for (int i = 0; i < 8; ++i) {
        int grow = row0 + wm * 32 + r * 16 + lhi * 8 + i;
        float v = acc[r][c][i];
        if (flags & FLAG_BIAS) v += bias[gcol];
        if (flags & FLAG_GELU) v = 0.5f * v * (1.f + erff(v * 0.70710678118f));
        if (flags & FLAG_RES)  v += resid[(size_t)grow * Nc + gcol];
        Cm[(size_t)grow * Nc + gcol] = v;
      }
    }
  }
}

// ---------------- weight pack: W[K,Nc] f32 -> Wt[Nc,K] f16 ----------------
__global__ __launch_bounds__(256)
void pack_w_kernel(const float* __restrict__ W, _Float16* __restrict__ Wt,
                   int K, int Nc) {
  __shared__ _Float16 tile[64][65];
  const int k0 = blockIdx.x * 64, n0 = blockIdx.y * 64;
  const int tid = threadIdx.x;
#pragma unroll
  for (int t = 0; t < 16; ++t) {
    int e = tid + t * 256;
    int r = e >> 6, c = e & 63;                  // r: k-offset, c: n-offset
    int k = k0 + r, n = n0 + c;
    float v = (n < Nc) ? W[(size_t)k * Nc + n] : 0.f;   // K always multiple of 64
    tile[r][c] = (_Float16)v;
  }
  __syncthreads();
#pragma unroll
  for (int t = 0; t < 16; ++t) {
    int e = tid + t * 256;
    int rr = e >> 6, cc = e & 63;                // rr: n-offset, cc: k-offset
    int n = n0 + rr, k = k0 + cc;
    if (n < Nc) Wt[(size_t)n * K + k] = tile[cc][rr];
  }
}

// ---------------- patch extraction + LN1 ----------------
__global__ __launch_bounds__(256)
void patch_ln_kernel(const float* __restrict__ img, const float* __restrict__ w,
                     const float* __restrict__ bv, float* __restrict__ out) {
  __shared__ float buf[PDIM];
  __shared__ float red[256];
  const int t = blockIdx.x;
  const int b = blockIdx.y;
  const int tid = threadIdx.x;
  const int gh = t >> 3, gw = t & 7;
  for (int e = tid; e < PDIM; e += 256) {
    int p12 = e / 3, ch = e % 3;
    int p1 = p12 >> 4, p2 = p12 & 15;
    buf[e] = img[(((size_t)b * 3 + ch) * 128 + gh * 16 + p1) * 128 + gw * 16 + p2];
  }
  __syncthreads();
  float s = 0.f;
  for (int e = tid; e < PDIM; e += 256) s += buf[e];
  red[tid] = s; __syncthreads();
  for (int st = 128; st > 0; st >>= 1) { if (tid < st) red[tid] += red[tid + st]; __syncthreads(); }
  float mean = red[0] * (1.f / PDIM);
  __syncthreads();
  float s2 = 0.f;
  for (int e = tid; e < PDIM; e += 256) { float d = buf[e] - mean; s2 += d * d; }
  red[tid] = s2; __syncthreads();
  for (int st = 128; st > 0; st >>= 1) { if (tid < st) red[tid] += red[tid + st]; __syncthreads(); }
  float rstd = rsqrtf(red[0] * (1.f / PDIM) + 1e-5f);
  float* op = out + ((size_t)b * 64 + t) * PDIM;
  for (int e = tid; e < PDIM; e += 256) op[e] = (buf[e] - mean) * rstd * w[e] + bv[e];
}

// ---------------- generic LayerNorm (strided input, compact output) -------
__global__ __launch_bounds__(256)
void ln_kernel(const float* __restrict__ in, const float* __restrict__ w,
               const float* __restrict__ bv, float* __restrict__ out,
               int width, long in_stride) {
  __shared__ float red[256];
  const int row = blockIdx.x, tid = threadIdx.x;
  const float* ip = in + (size_t)row * in_stride;
  float s = 0.f;
  for (int i = tid; i < width; i += 256) s += ip[i];
  red[tid] = s; __syncthreads();
  for (int st = 128; st > 0; st >>= 1) { if (tid < st) red[tid] += red[tid + st]; __syncthreads(); }
  float mean = red[0] / width;
  __syncthreads();
  float s2 = 0.f;
  for (int i = tid; i < width; i += 256) { float d = ip[i] - mean; s2 += d * d; }
  red[tid] = s2; __syncthreads();
  for (int st = 128; st > 0; st >>= 1) { if (tid < st) red[tid] += red[tid + st]; __syncthreads(); }
  float rstd = rsqrtf(red[0] / width + 1e-5f);
  __syncthreads();
  for (int i = tid; i < width; i += 256)
    out[(size_t)row * width + i] = (ip[i] - mean) * rstd * w[i] + bv[i];
}

// ---------------- cls + pos-emb assembly ----------------
__global__ void assemble_kernel(const float* __restrict__ pe, const float* __restrict__ cls,
                                const float* __restrict__ pos, float* __restrict__ x) {
  const int t = blockIdx.x, b = blockIdx.y, tid = threadIdx.x;
  float* xp = x + ((size_t)b * NTOK + t) * DIMD;
  const float* pp = pos + (size_t)t * DIMD;
  if (t == 0) {
    for (int i = tid; i < DIMD; i += blockDim.x) xp[i] = cls[i] + pp[i];
  } else {
    const float* pr = pe + ((size_t)b * 64 + (t - 1)) * DIMD;
    for (int i = tid; i < DIMD; i += blockDim.x) xp[i] = pr[i] + pp[i];
  }
}

// ---------------- attention: dots + rel-bias + top-16 softmax + attn@V ----
__global__ __launch_bounds__(128)
void attn_topk_kernel(const float* __restrict__ qkv, const float* __restrict__ btab,
                      float* __restrict__ o) {
  const int b = blockIdx.z, h = blockIdx.y, qi = blockIdx.x;
  const int tid = threadIdx.x;   // 128
  __shared__ float qs[DHEAD];
  __shared__ float dots[NTOK];
  __shared__ float rv[128];
  __shared__ int   ri[128];
  __shared__ unsigned char sel[NTOK];
  __shared__ float s_max, s_z;

  if (tid < DHEAD) qs[tid] = qkv[((size_t)(b * NTOK + qi)) * QKVW + h * DHEAD + tid];
  if (tid < NTOK) sel[tid] = 0;
  __syncthreads();

  for (int j = tid; j < NTOK; j += 128) {
    const float* kr = &qkv[((size_t)(b * NTOK + j)) * QKVW + INNER + h * DHEAD];
    float s = 0.f;
    for (int d = 0; d < DHEAD; ++d) s += qs[d] * kr[d];
    float bias = 0.f;
    if (qi > 0 && j > 0) {
      int pi = qi - 1, pj = j - 1;
      int dr = (pi >> 3) - (pj >> 3) + 7;
      int dc = (pi & 7) - (pj & 7) + 7;
      bias = btab[dr * 15 + dc];
    }
    dots[j] = s * 0.125f + bias;
  }
  __syncthreads();

  for (int it = 0; it < TOPK; ++it) {
    float v = (tid < NTOK && !sel[tid]) ? dots[tid] : -3.402823466e38f;
    rv[tid] = v; ri[tid] = tid;
    __syncthreads();
    for (int st = 64; st > 0; st >>= 1) {
      if (tid < st && rv[tid + st] > rv[tid]) { rv[tid] = rv[tid + st]; ri[tid] = ri[tid + st]; }
      __syncthreads();
    }
    if (tid == 0) { sel[ri[0]] = 1; if (it == 0) s_max = rv[0]; }
    __syncthreads();
  }

  float wgt = 0.f;
  if (tid < NTOK && sel[tid]) wgt = __expf(dots[tid] - s_max);
  rv[tid] = wgt; __syncthreads();
  for (int st = 64; st > 0; st >>= 1) { if (tid < st) rv[tid] += rv[tid + st]; __syncthreads(); }
  if (tid == 0) s_z = rv[0];
  __syncthreads();
  if (tid < NTOK) dots[tid] = wgt / s_z;
  __syncthreads();

  if (tid < DHEAD) {
    float acc = 0.f;
    for (int j = 0; j < NTOK; ++j)
      acc += dots[j] * qkv[((size_t)(b * NTOK + j)) * QKVW + 2 * INNER + h * DHEAD + tid];
    o[((size_t)(b * NTOK + qi)) * INNER + h * DHEAD + tid] = acc;
  }
}

// ---------------- host orchestration ----------------
extern "C" void kernel_launch(void* const* d_in, const int* in_sizes, int n_in,
                              void* d_out, int out_size, void* d_ws, size_t ws_size,
                              hipStream_t stream) {
  (void)in_sizes; (void)n_in; (void)out_size; (void)ws_size;

  const float* img      = (const float*)d_in[0];
  const float* pe_ln1_w = (const float*)d_in[1];
  const float* pe_ln1_b = (const float*)d_in[2];
  const float* pe_w     = (const float*)d_in[3];
  const float* pe_b     = (const float*)d_in[4];
  const float* pe_ln2_w = (const float*)d_in[5];
  const float* pe_ln2_b = (const float*)d_in[6];
  const float* cls_tok  = (const float*)d_in[7];
  const float* pos_emb  = (const float*)d_in[8];
  const float* attn_ln_w = (const float*)d_in[9];
  const float* attn_ln_b = (const float*)d_in[10];
  const float* qkv_w     = (const float*)d_in[11];
  const float* out_w     = (const float*)d_in[12];
  const float* out_b     = (const float*)d_in[13];
  const float* bias_tab  = (const float*)d_in[14];
  const float* ff_ln_w   = (const float*)d_in[15];
  const float* ff_ln_b   = (const float*)d_in[16];
  const float* ff_w1     = (const float*)d_in[17];
  const float* ff_b1     = (const float*)d_in[18];
  const float* ff_w2     = (const float*)d_in[19];
  const float* ff_b2     = (const float*)d_in[20];
  const float* fin_ln_w  = (const float*)d_in[21];
  const float* fin_ln_b  = (const float*)d_in[22];
  const float* head_w    = (const float*)d_in[23];
  const float* head_b    = (const float*)d_in[24];
  float* out = (float*)d_out;

  const size_t M = (size_t)BATCH * NTOK;              // 33280 (multiple of 128)
  float* ws   = (float*)d_ws;
  float* x    = ws;                                   // [M, 384]
  float* hbuf = x    + M * DIMD;                      // [M, 384]
  float* big  = hbuf + M * DIMD;                      // [M, 1536] region
  float* qkvb = big;                                  // [M, 1152]
  float* obuf = big + M * QKVW;                       // [M, 384]
  float* ffb  = big;                                  // [M, 1536] (reuses qkv+o)
  float* patches = big;                               // [B*64, 768] (pre-layer)
  float* peout   = obuf;                              // [B*64, 384] (disjoint)
  float* clsbuf  = hbuf;                              // [B, 384]

  // f16 packed-weight region after the float buffers
  _Float16* wbase  = (_Float16*)(big + M * (size_t)MLPD);
  _Float16* wt_pe  = wbase;                                   // 384*768
  _Float16* wt_hd  = wt_pe + (size_t)DIMD * PDIM;             // 1000*384
  _Float16* wt_lyr = wt_hd + (size_t)NCLS * DIMD;             // per-layer block
  const size_t L_QKV = (size_t)QKVW * DIMD;                   // 442368
  const size_t L_OUT = (size_t)DIMD * INNER;                  // 147456
  const size_t L_FF1 = (size_t)MLPD * DIMD;                   // 589824
  const size_t L_FF2 = (size_t)DIMD * MLPD;                   // 589824
  const size_t L_ALL = L_QKV + L_OUT + L_FF1 + L_FF2;

  const int MROWS = (int)M;

  // ---- pack all weights to f16 transposed [Nc][K] ----
  pack_w_kernel<<<dim3(PDIM / 64, DIMD / 64), 256, 0, stream>>>(pe_w, wt_pe, PDIM, DIMD);
  pack_w_kernel<<<dim3(DIMD / 64, (NCLS + 63) / 64), 256, 0, stream>>>(head_w, wt_hd, DIMD, NCLS);
  for (int i = 0; i < DEPTH; ++i) {
    _Float16* wl = wt_lyr + (size_t)i * L_ALL;
    pack_w_kernel<<<dim3(DIMD / 64, QKVW / 64), 256, 0, stream>>>(
        qkv_w + (size_t)i * DIMD * QKVW, wl, DIMD, QKVW);
    pack_w_kernel<<<dim3(INNER / 64, DIMD / 64), 256, 0, stream>>>(
        out_w + (size_t)i * INNER * DIMD, wl + L_QKV, INNER, DIMD);
    pack_w_kernel<<<dim3(DIMD / 64, MLPD / 64), 256, 0, stream>>>(
        ff_w1 + (size_t)i * DIMD * MLPD, wl + L_QKV + L_OUT, DIMD, MLPD);
    pack_w_kernel<<<dim3(MLPD / 64, DIMD / 64), 256, 0, stream>>>(
        ff_w2 + (size_t)i * MLPD * DIMD, wl + L_QKV + L_OUT + L_FF1, MLPD, DIMD);
  }

  // ---- patch embedding ----
  patch_ln_kernel<<<dim3(64, BATCH), 256, 0, stream>>>(img, pe_ln1_w, pe_ln1_b, patches);
  gemm_wmma_kernel<false><<<dim3(32768 / BM, DIMD / BN), 256, 0, stream>>>(
      patches, wt_pe, pe_b, nullptr, peout, 32768, PDIM, DIMD, FLAG_BIAS);
  ln_kernel<<<32768, 256, 0, stream>>>(peout, pe_ln2_w, pe_ln2_b, peout, DIMD, DIMD);
  assemble_kernel<<<dim3(NTOK, BATCH), 128, 0, stream>>>(peout, cls_tok, pos_emb, x);

  // ---- transformer layers ----
  for (int i = 0; i < DEPTH; ++i) {
    _Float16* wl = wt_lyr + (size_t)i * L_ALL;
    ln_kernel<<<MROWS, 256, 0, stream>>>(x, attn_ln_w + (size_t)i * DIMD,
                                         attn_ln_b + (size_t)i * DIMD, hbuf, DIMD, DIMD);
    gemm_wmma_kernel<false><<<dim3(MROWS / BM, QKVW / BN), 256, 0, stream>>>(
        hbuf, wl, nullptr, nullptr, qkvb, MROWS, DIMD, QKVW, 0);
    attn_topk_kernel<<<dim3(NTOK, HEADS, BATCH), 128, 0, stream>>>(
        qkvb, bias_tab + (size_t)i * 225, obuf);
    gemm_wmma_kernel<false><<<dim3(MROWS / BM, DIMD / BN), 256, 0, stream>>>(
        obuf, wl + L_QKV, out_b + (size_t)i * DIMD, x, x,
        MROWS, INNER, DIMD, FLAG_BIAS | FLAG_RES);
    ln_kernel<<<MROWS, 256, 0, stream>>>(x, ff_ln_w + (size_t)i * DIMD,
                                         ff_ln_b + (size_t)i * DIMD, hbuf, DIMD, DIMD);
    gemm_wmma_kernel<false><<<dim3(MROWS / BM, MLPD / BN), 256, 0, stream>>>(
        hbuf, wl + L_QKV + L_OUT, ff_b1 + (size_t)i * MLPD, nullptr, ffb,
        MROWS, DIMD, MLPD, FLAG_BIAS | FLAG_GELU);
    gemm_wmma_kernel<false><<<dim3(MROWS / BM, DIMD / BN), 256, 0, stream>>>(
        ffb, wl + L_QKV + L_OUT + L_FF1, ff_b2 + (size_t)i * DIMD, x, x,
        MROWS, MLPD, DIMD, FLAG_BIAS | FLAG_RES);
  }

  // ---- final LN on cls rows + classifier head ----
  ln_kernel<<<BATCH, 256, 0, stream>>>(x, fin_ln_w, fin_ln_b, clsbuf, DIMD, (long)NTOK * DIMD);
  gemm_wmma_kernel<true><<<dim3(BATCH / BM, (NCLS + BN - 1) / BN), 256, 0, stream>>>(
      clsbuf, wt_hd, head_b, nullptr, out, BATCH, DIMD, NCLS, FLAG_BIAS);
}
